// NBFNet_3745211482944
// MI455X (gfx1250) — matrix operand order repaired and checked by reading, not compiled, not run on hardware
//
#include <hip/hip_runtime.h>
#include <hip/hip_bf16.h>

#define NE      50000
#define DIM     64
#define NREL    400
#define NLAYERS 3
#define BS      4
#define NEDGE   400000

#define NTILES_E (NEDGE / 16)          // 25000 edge tiles
#define NROWS    (BS * NE)             // 200000
#define NTILES_U (NROWS / 16)          // 12500

typedef __attribute__((ext_vector_type(16))) __bf16 v16bf;
typedef __attribute__((ext_vector_type(8)))  __bf16 v8bf;
typedef __attribute__((ext_vector_type(8)))  float  v8f;

__device__ __forceinline__ v8f wmma_bf16(v16bf a, v16bf b, v8f c) {
    return __builtin_amdgcn_wmma_f32_16x16x32_bf16(false, a, false, b, (short)0, c,
                                                   false, false);
}

__device__ __forceinline__ float fast_sigmoid(float x) {
    return __builtin_amdgcn_rcpf(1.0f + __expf(-x));
}

// Stuff 8 floats (two float4 b128 loads) into A-fragment elements [off..off+8)
template <int OFF>
__device__ __forceinline__ void cvt8(v16bf& A, float4 a, float4 b) {
    A[OFF + 0] = (__bf16)a.x; A[OFF + 1] = (__bf16)a.y;
    A[OFF + 2] = (__bf16)a.z; A[OFF + 3] = (__bf16)a.w;
    A[OFF + 4] = (__bf16)b.x; A[OFF + 5] = (__bf16)b.y;
    A[OFF + 6] = (__bf16)b.z; A[OFF + 7] = (__bf16)b.w;
}

// A fragment (16-bit A 16x32 layout: lanes 0-15 take K = 0..7 / 16..23,
// lanes 16-31 take K = 8..15 / 24..31) from a 32-float fp32 stretch.
__device__ __forceinline__ v16bf load_a_f32(const float* __restrict__ row32,
                                            int half) {
    const float4* r4 = (const float4*)row32;
    int b = half * 2;                  // half*8 floats
    float4 p0 = r4[b + 0];
    float4 p1 = r4[b + 1];
    float4 p2 = r4[b + 4];             // +16 floats
    float4 p3 = r4[b + 5];
    v16bf A;
    cvt8<0>(A, p0, p1);
    cvt8<8>(A, p2, p3);
    return A;
}

// Same fragment from a 32-element bf16 stretch: two 16-byte loads, no cvt.
__device__ __forceinline__ v16bf load_a_bf(const __bf16* __restrict__ row32,
                                           int half) {
    const v8bf* r = (const v8bf*)(row32 + half * 8);
    v8bf lo = r[0];                    // K = half*8 .. +7
    v8bf hi = r[2];                    // K = 16 + half*8 .. +7
    return __builtin_shufflevector(lo, hi, 0, 1, 2, 3, 4, 5, 6, 7,
                                   8, 9, 10, 11, 12, 13, 14, 15);
}

// ---------------------------------------------------------------------------
// 0. pack: convert a (KT*32 x 64) fp32 weight matrix into bf16 B fragments.
//    Fragment (kc,nc) for lane L lives at pack[((kc*4+nc)*32+L)*16 .. +16):
//    element j -> W[kc*32 + (j<8 ? (L>>4)*8 + j : 16 + (L>>4)*8 + j-8), nc*16 + (L&15)]   (ISA B 32x16)
// ---------------------------------------------------------------------------
__global__ void pack_w_kernel(const float* __restrict__ W,
                              __bf16* __restrict__ out, int KT) {
    int idx = blockIdx.x * blockDim.x + threadIdx.x;
    if (idx >= KT * 4 * 32) return;
    int lane = idx & 31;
    int kcnc = idx >> 5;
    int kc = kcnc >> 2, nc = kcnc & 3;
    int m = lane & 15, half = lane >> 4;
    int n = nc * 16 + m;
    __bf16* o = out + (size_t)idx * 16;
#pragma unroll
    for (int j = 0; j < 16; ++j) {
        int K = kc * 32 + (j < 8 ? half * 8 + j : 16 + half * 8 + (j - 8));
        o[j] = (__bf16)W[K * DIM + n];
    }
}

// ---------------------------------------------------------------------------
// 1. init: h[b, source[b], :] = ent_emb[source[b]] + query_emb[query_rel[b]]
// ---------------------------------------------------------------------------
__global__ void init_kernel(const float* __restrict__ ent,
                            const float* __restrict__ qemb,
                            const int* __restrict__ source,
                            const int* __restrict__ qrel,
                            __bf16* __restrict__ h) {
    int b = threadIdx.x >> 6;      // 256 threads = BS * DIM
    int n = threadIdx.x & 63;
    int s = source[b];
    int q = qrel[b];
    h[((size_t)b * NE + s) * DIM + n] =
        (__bf16)(ent[(size_t)s * DIM + n] + qemb[(size_t)q * DIM + n]);
}

// ---------------------------------------------------------------------------
// 2. edge message: one wave = one (16-edge tile, batch) pair.
//    gate = sigmoid(h_src @ gate_W); msg = (h_src @ msg_W) * r * gate
//    atomicAdd (fp32) into h_agg[b, edge_tgt].
//    Addresses: per-row byte offsets hoisted once (32-bit), per-nc base bump.
// ---------------------------------------------------------------------------
__global__ void edge_msg_kernel(const __bf16* __restrict__ h,
                                float* __restrict__ hagg,
                                const v16bf* __restrict__ packG,   // bf16 frags
                                const v16bf* __restrict__ packM,
                                const float* __restrict__ relE,    // NREL x 64
                                const int* __restrict__ edge_src,
                                const int* __restrict__ edge_tgt,
                                const int* __restrict__ edge_rel) {
    int wave = threadIdx.x >> 5;
    int lane = threadIdx.x & 31;
    int tile = blockIdx.x * 8 + wave;          // over NTILES_E * BS
    int b    = tile & 3;
    int et   = tile >> 2;
    if (et >= NTILES_E) return;

    int m    = lane & 15;
    int half = lane >> 4;
    int e0   = et * 16;

    int src_m = edge_src[e0 + m];          // row this lane gathers for A

    const unsigned ROWB = DIM * sizeof(float);   // 256 bytes per fp32 row
    unsigned toff[8], roff[8];             // byte offsets of this lane's D rows
#pragma unroll
    for (int v = 0; v < 8; ++v) {
        toff[v] = (unsigned)edge_tgt[e0 + half * 8 + v] * ROWB;
        roff[v] = (unsigned)edge_rel[e0 + half * 8 + v] * ROWB;
    }

    // A fragments: gathered bf16 h_src row for this batch (K = 64 -> 2x32)
    const __bf16* hr = h + ((size_t)b * NE + src_m) * DIM;
    v16bf A0 = load_a_bf(hr, half);
    v16bf A1 = load_a_bf(hr + 32, half);

    char* haggB       = (char*)(hagg + (size_t)b * NE * DIM);
    const char* relB  = (const char*)relE;

#pragma unroll
    for (int nc = 0; nc < 4; ++nc) {
        int n = nc * 16 + m;
        v16bf Bg0 = packG[(0 * 4 + nc) * 32 + lane];
        v16bf Bg1 = packG[(1 * 4 + nc) * 32 + lane];
        v16bf Bm0 = packM[(0 * 4 + nc) * 32 + lane];
        v16bf Bm1 = packM[(1 * 4 + nc) * 32 + lane];

        v8f g = {}, s = {};
        g = wmma_bf16(A0, Bg0, g);
        g = wmma_bf16(A1, Bg1, g);
        s = wmma_bf16(A0, Bm0, s);
        s = wmma_bf16(A1, Bm1, s);

        char* haggBn      = haggB + (unsigned)(n * sizeof(float));
        const char* relBn = relB + (unsigned)(n * sizeof(float));
#pragma unroll
        for (int v = 0; v < 8; ++v) {
            float rv  = *(const float*)(relBn + roff[v]);
            float msg = s[v] * rv * fast_sigmoid(g[v]);
            atomicAdd((float*)(haggBn + toff[v]), msg);
        }
    }
}

// ---------------------------------------------------------------------------
// 3. update: pre = h + relu(concat(h, h_agg) @ upd_W + upd_b) ; writes h_agg
// ---------------------------------------------------------------------------
__global__ void update_kernel(const __bf16* __restrict__ h,
                              float* __restrict__ hagg,   // in: agg, out: pre-LN
                              const v16bf* __restrict__ packU,  // 128x64 frags
                              const float* __restrict__ updb) { // 64
    int wave = threadIdx.x >> 5;
    int lane = threadIdx.x & 31;
    int tile = blockIdx.x * 8 + wave;
    if (tile >= NTILES_U) return;

    int m    = lane & 15;
    int half = lane >> 4;
    size_t rowA = (size_t)(tile * 16 + m) * DIM;

    v16bf A[4];                             // K = 128: [h | h_agg]
    A[0] = load_a_bf(h + rowA, half);
    A[1] = load_a_bf(h + rowA + 32, half);
    A[2] = load_a_f32(hagg + rowA, half);
    A[3] = load_a_f32(hagg + rowA + 32, half);

#pragma unroll
    for (int nc = 0; nc < 4; ++nc) {
        int n = nc * 16 + m;
        v8f acc = {};
#pragma unroll
        for (int kc = 0; kc < 4; ++kc)
            acc = wmma_bf16(A[kc], packU[(kc * 4 + nc) * 32 + lane], acc);

        float bias = updb[n];
#pragma unroll
        for (int v = 0; v < 8; ++v) {
            size_t row = (size_t)(tile * 16 + half * 8 + v) * DIM;
            float u = acc[v] + bias;
            u = u > 0.0f ? u : 0.0f;
            hagg[row + n] = (float)h[row + n] + u;
        }
    }
}

// ---------------------------------------------------------------------------
// 4. LayerNorm: h = LN(pre) * g + b   (wave per row, wave32 shfl reductions)
//    fp32 statistics, bf16 output.
// ---------------------------------------------------------------------------
__global__ void ln_kernel(const float* __restrict__ pre,
                          __bf16* __restrict__ h,
                          const float* __restrict__ g,
                          const float* __restrict__ b) {
    int row  = blockIdx.x * 8 + (threadIdx.x >> 5);
    int lane = threadIdx.x & 31;
    const float* x = pre + (size_t)row * DIM;
    float x0 = x[lane];
    float x1 = x[lane + 32];

    float s = x0 + x1;
#pragma unroll
    for (int off = 16; off > 0; off >>= 1) s += __shfl_xor(s, off, 32);
    float mean = s * (1.0f / DIM);

    float d0 = x0 - mean, d1 = x1 - mean;
    float vs = d0 * d0 + d1 * d1;
#pragma unroll
    for (int off = 16; off > 0; off >>= 1) vs += __shfl_xor(vs, off, 32);
    float rstd = rsqrtf(vs * (1.0f / DIM) + 1e-5f);

    __bf16* y = h + (size_t)row * DIM;
    y[lane]      = (__bf16)(d0 * rstd * g[lane]      + b[lane]);
    y[lane + 32] = (__bf16)(d1 * rstd * g[lane + 32] + b[lane + 32]);
}

// ---------------------------------------------------------------------------
// 5. score: out = relu(concat(h, ent) @ sW1 + sb1) @ sW2 + sb2
// ---------------------------------------------------------------------------
__global__ void score_kernel(const __bf16* __restrict__ h,
                             const float* __restrict__ ent,
                             const v16bf* __restrict__ packS,  // 128x64 frags
                             const float* __restrict__ sb1,    // 64
                             const float* __restrict__ sW2,    // 64
                             const float* __restrict__ sb2,    // 1
                             float* __restrict__ out) {
    int wave = threadIdx.x >> 5;
    int lane = threadIdx.x & 31;
    int tile = blockIdx.x * 8 + wave;
    if (tile >= NTILES_U) return;

    int m    = lane & 15;
    int half = lane >> 4;
    int row_m = tile * 16 + m;
    int e_m   = row_m % NE;

    v16bf A[4];                             // K = 128: [h | ent_emb]
    A[0] = load_a_bf(h + (size_t)row_m * DIM, half);
    A[1] = load_a_bf(h + (size_t)row_m * DIM + 32, half);
    A[2] = load_a_f32(ent + (size_t)e_m * DIM, half);
    A[3] = load_a_f32(ent + (size_t)e_m * DIM + 32, half);

    float part[8] = {0, 0, 0, 0, 0, 0, 0, 0};
#pragma unroll
    for (int nc = 0; nc < 4; ++nc) {
        int n = nc * 16 + m;
        v8f acc = {};
#pragma unroll
        for (int kc = 0; kc < 4; ++kc)
            acc = wmma_bf16(A[kc], packS[(kc * 4 + nc) * 32 + lane], acc);

        float bias = sb1[n];
        float w2   = sW2[n];
#pragma unroll
        for (int v = 0; v < 8; ++v) {
            float s1 = acc[v] + bias;
            s1 = s1 > 0.0f ? s1 : 0.0f;
            part[v] += s1 * w2;
        }
    }
    // reduce over the 16 feature-lanes of each half (rows stay in their half)
#pragma unroll
    for (int off = 8; off > 0; off >>= 1)
#pragma unroll
        for (int v = 0; v < 8; ++v) part[v] += __shfl_xor(part[v], off, 32);

    if (m == 0) {
        float s2 = sb2[0];
#pragma unroll
        for (int v = 0; v < 8; ++v)
            out[tile * 16 + half * 8 + v] = part[v] + s2;
    }
}

// ---------------------------------------------------------------------------
extern "C" void kernel_launch(void* const* d_in, const int* in_sizes, int n_in,
                              void* d_out, int out_size, void* d_ws, size_t ws_size,
                              hipStream_t stream) {
    const float* ent_emb   = (const float*)d_in[0];
    const float* query_emb = (const float*)d_in[1];
    const float* rel_embs  = (const float*)d_in[2];
    const float* msg_W     = (const float*)d_in[3];
    const float* gate_W    = (const float*)d_in[4];
    const float* upd_W     = (const float*)d_in[5];
    const float* upd_b     = (const float*)d_in[6];
    const float* ln_g      = (const float*)d_in[7];
    const float* ln_b      = (const float*)d_in[8];
    const float* sW1       = (const float*)d_in[9];
    const float* sb1       = (const float*)d_in[10];
    const float* sW2       = (const float*)d_in[11];
    const float* sb2       = (const float*)d_in[12];
    const int* source      = (const int*)d_in[13];
    const int* query_rel   = (const int*)d_in[14];
    const int* edge_src    = (const int*)d_in[15];
    const int* edge_tgt    = (const int*)d_in[16];
    const int* edge_rel    = (const int*)d_in[17];

    const size_t HN   = (size_t)BS * NE * DIM;    // 12.8M elements
    const size_t W64  = 2 * 4 * 32 * 16;          // 4096 bf16 per 64x64 matrix
    const size_t W128 = 4 * 4 * 32 * 16;          // 8192 bf16 per 128x64 matrix

    __bf16* h   = (__bf16*)d_ws;                  // HN bf16 (25.6 MB)
    float* hagg = (float*)(h + HN);               // HN fp32 (51.2 MB)
    __bf16* pkGate = (__bf16*)(hagg + HN);        // 3 * W64
    __bf16* pkMsg  = pkGate + NLAYERS * W64;      // 3 * W64
    __bf16* pkUpd  = pkMsg + NLAYERS * W64;       // 3 * W128
    __bf16* pkS    = pkUpd + NLAYERS * W128;      // 1 * W128

    // one-time bf16 weight packing into WMMA B-fragment layout
    for (int l = 0; l < NLAYERS; ++l) {
        pack_w_kernel<<<1, 256, 0, stream>>>(gate_W + (size_t)l * DIM * DIM,
                                             pkGate + (size_t)l * W64, 2);
        pack_w_kernel<<<1, 256, 0, stream>>>(msg_W + (size_t)l * DIM * DIM,
                                             pkMsg + (size_t)l * W64, 2);
        pack_w_kernel<<<2, 256, 0, stream>>>(upd_W + (size_t)l * 2 * DIM * DIM,
                                             pkUpd + (size_t)l * W128, 4);
    }
    pack_w_kernel<<<2, 256, 0, stream>>>(sW1, pkS, 4);

    hipMemsetAsync(h, 0, HN * sizeof(__bf16), stream);
    init_kernel<<<1, BS * DIM, 0, stream>>>(ent_emb, query_emb, source, query_rel, h);

    for (int l = 0; l < NLAYERS; ++l) {
        hipMemsetAsync(hagg, 0, HN * sizeof(float), stream);
        edge_msg_kernel<<<(NTILES_E * BS) / 8, 256, 0, stream>>>(
            h, hagg,
            (const v16bf*)(pkGate + (size_t)l * W64),
            (const v16bf*)(pkMsg + (size_t)l * W64),
            rel_embs + (size_t)l * NREL * DIM,
            edge_src, edge_tgt, edge_rel);
        update_kernel<<<(NTILES_U + 7) / 8, 256, 0, stream>>>(
            h, hagg, (const v16bf*)(pkUpd + (size_t)l * W128),
            upd_b + (size_t)l * DIM);
        ln_kernel<<<NROWS / 8, 256, 0, stream>>>(
            hagg, h, ln_g + (size_t)l * DIM, ln_b + (size_t)l * DIM);
    }

    score_kernel<<<(NTILES_U + 7) / 8, 256, 0, stream>>>(
        h, ent_emb, (const v16bf*)pkS, sb1, sW2, sb2, (float*)d_out);
}
